// _CausalSelfAttention_50010599194882
// MI455X (gfx1250) — matrix operand order repaired and checked by reading, not compiled, not run on hardware
//
#include <hip/hip_runtime.h>
#include <hip/hip_bf16.h>
#include <stdint.h>

// ---------------------------------------------------------------------------
// CDNA5 (gfx1250) causal self-attention, bf16 WMMA path + TDM-staged GEMM.
//   B=4, T=2048, C=1024, H=16, D=64
// Phases: f32->bf16 convert, QKV GEMM (TDM B-tiles), V transpose,
//         flash attention, proj GEMM (TDM B-tiles).
// ---------------------------------------------------------------------------

typedef __attribute__((ext_vector_type(16))) __bf16 v16bf;
typedef __attribute__((ext_vector_type(8)))  float  v8f;
typedef __attribute__((ext_vector_type(4)))  unsigned int v4u;
typedef __attribute__((ext_vector_type(8)))  int    v8i;
typedef __attribute__((ext_vector_type(4)))  int    v4i;

union FragU {
  v16bf v;
  uint4 q[2];
};

// Build a 16-element bf16 fragment from two 16B chunks at p and p+off2 elems.
__device__ __forceinline__ v16bf ld_frag(const unsigned short* p, int off2) {
  FragU u;
  u.q[0] = *(const uint4*)(p);
  u.q[1] = *(const uint4*)(p + off2);
  return u.v;
}

__device__ __forceinline__ unsigned short f2bf(float f) {
  unsigned int u = __float_as_uint(f);
  u += 0x7fffu + ((u >> 16) & 1u);   // round to nearest even
  return (unsigned short)(u >> 16);
}

__device__ __forceinline__ v8f wmma_bf16(v16bf a, v16bf b, v8f c) {
  // v_wmma_f32_16x16x32_bf16: (neg_a, A, neg_b, B, c_mod, C, reuse_a, reuse_b)
  return __builtin_amdgcn_wmma_f32_16x16x32_bf16(false, a, false, b, (short)0, c,
                                                 false, false);
}

// ---------------------------------------------------------------------------
// Tensor Data Mover: DMA a 2-D bf16 tile [tile_d1 rows x tile_d0 elems] from
// global (row stride = stride0 elems) into LDS at lds_off (rows packed,
// row stride = tile_d0 elems). D# layout per CDNA5 ISA ch.8.
__device__ __forceinline__ void tdm_load_2d(unsigned int lds_off,
                                            const unsigned short* gptr,
                                            unsigned int tensor_d0,
                                            unsigned int tensor_d1,
                                            unsigned int stride0,
                                            unsigned int tile_d0,
                                            unsigned int tile_d1) {
  unsigned long long ga = (unsigned long long)(uintptr_t)gptr;
  v4u g0;
  g0.x = 1u;                                   // count=1, user mode, no gather
  g0.y = lds_off;                              // lds_addr (bytes)
  g0.z = (unsigned int)(ga & 0xFFFFFFFFu);     // global_addr[31:0]
  g0.w = (unsigned int)((ga >> 32) & 0x01FFFFFFu) | (2u << 30);  // [56:32]|type=2
  v8i g1;
  g1[0] = (int)(1u << 16);                                       // data_size=2B
  g1[1] = (int)((tensor_d0 & 0xFFFFu) << 16);                    // abar=0|dim0.lo
  g1[2] = (int)((tensor_d0 >> 16) | ((tensor_d1 & 0xFFFFu) << 16));
  g1[3] = (int)((tensor_d1 >> 16) | (tile_d0 << 16));
  g1[4] = (int)(tile_d1);                                        // tile_dim2=0
  g1[5] = (int)stride0;                                          // dim0_stride lo
  g1[6] = 0;                                                     // stride hi bits
  g1[7] = 0;
  v4i z4 = (v4i)(0);
#if defined(__clang_major__) && (__clang_major__ >= 23)
  v8i z8 = (v8i)(0);
  __builtin_amdgcn_tensor_load_to_lds(g0, g1, z4, z4, z8, 0);
#else
  __builtin_amdgcn_tensor_load_to_lds(g0, g1, z4, z4, 0);
#endif
}

// ---------------------------------------------------------------------------
__global__ void cvt_f32_bf16(const float* __restrict__ in,
                             unsigned short* __restrict__ out, int n) {
  int i = blockIdx.x * 256 + threadIdx.x;
  if (i < n) out[i] = f2bf(in[i]);
}

// qkv[b,t, 2048 + h*64 + d] -> vT[((b*16+h)*64 + d)*2048 + t]
__global__ void vtrans_kernel(const unsigned short* __restrict__ qkv,
                              unsigned short* __restrict__ vT) {
  int i = blockIdx.x * 256 + threadIdx.x;  // exactly 8388608 threads
  int t = i & 2047;
  int d = (i >> 11) & 63;
  int h = (i >> 17) & 15;
  int b = i >> 21;
  vT[i] = qkv[(size_t)(b * 2048 + t) * 3072 + 2048 + h * 64 + d];
}

// ---------------------------------------------------------------------------
// C[M,N] = A[M,K] * B[N,K]^T  (both bf16 row-major, K contiguous).
// Block = 4 waves; each wave computes a 32x64 tile (2 Mt x 4 Nt of 16x16).
// Workgroup tile: 64(M) x 128(N).  B tile (128 rows x 32 k) staged into LDS
// by the Tensor Data Mover, double-buffered, one DMA per workgroup per step.
template <bool F32OUT>
__global__ __launch_bounds__(128) void gemm_bf16nt(
    const unsigned short* __restrict__ A, const unsigned short* __restrict__ B,
    unsigned short* __restrict__ Cb, float* __restrict__ Cf,
    int K, int N, int lda, int ldb, int ldc) {
  __shared__ __align__(16) unsigned short sB[2][128 * 32];  // 2 x 8 KiB

  const int lane = threadIdx.x & 31;
  const int w    = threadIdx.x >> 5;
  const int ln   = lane & 15;
  const int hi   = lane >> 4;
  const int mB   = blockIdx.y * 64 + (w >> 1) * 32;
  const int nB0  = blockIdx.x * 128;            // workgroup N base (B tile)
  const int nW   = (w & 1) * 64;                // wave offset inside B tile

  v8f acc[2][4];
#pragma unroll
  for (int mt = 0; mt < 2; ++mt)
#pragma unroll
    for (int nt = 0; nt < 4; ++nt) acc[mt][nt] = (v8f)(0.0f);

  const unsigned short* arow[2];
  arow[0] = A + (size_t)(mB + ln) * lda;
  arow[1] = A + (size_t)(mB + 16 + ln) * lda;
  const unsigned short* btile = B + (size_t)nB0 * ldb;

  const int KT = K >> 5;
  if (w == 0)
    tdm_load_2d((unsigned int)(uintptr_t)&sB[0][0], btile,
                (unsigned)K, (unsigned)N, (unsigned)ldb, 32u, 128u);

  for (int kt = 0; kt < KT; ++kt) {
    const int k   = kt << 5;
    const int cur = kt & 1;
    if (w == 0) __builtin_amdgcn_s_wait_tensorcnt(0);
    __syncthreads();  // B tile 'cur' visible to all waves
    if (w == 0 && kt + 1 < KT)
      tdm_load_2d((unsigned int)(uintptr_t)&sB[cur ^ 1][0], btile + (k + 32),
                  (unsigned)K, (unsigned)N, (unsigned)ldb, 32u, 128u);

    // A-fragment: lane row = lane%16; K chunks {c0..c0+7, c0+16..c0+23}, c0=8*hi
    const int ca = k + hi * 8;
    v16bf a0 = ld_frag(arow[0] + ca, 16);
    v16bf a1 = ld_frag(arow[1] + ca, 16);
    // B-fragment from LDS: lane col = lane%16; K chunks {c0..c0+7, c0+16..c0+23}, c0=8*hi
    const unsigned short* bb = &sB[cur][0];
    v16bf bfr[4];
#pragma unroll
    for (int nt = 0; nt < 4; ++nt)
      bfr[nt] = ld_frag(bb + (nW + nt * 16 + ln) * 32 + hi * 8, 16);

    acc[0][0] = wmma_bf16(a0, bfr[0], acc[0][0]);
    acc[0][1] = wmma_bf16(a0, bfr[1], acc[0][1]);
    acc[0][2] = wmma_bf16(a0, bfr[2], acc[0][2]);
    acc[0][3] = wmma_bf16(a0, bfr[3], acc[0][3]);
    acc[1][0] = wmma_bf16(a1, bfr[0], acc[1][0]);
    acc[1][1] = wmma_bf16(a1, bfr[1], acc[1][1]);
    acc[1][2] = wmma_bf16(a1, bfr[2], acc[1][2]);
    acc[1][3] = wmma_bf16(a1, bfr[3], acc[1][3]);

    __syncthreads();  // all waves done reading 'cur' before it is re-filled
  }

  // C/D layout: VGPR r, lane L -> M = r + 8*(L/16), N = L%16
#pragma unroll
  for (int mt = 0; mt < 2; ++mt)
#pragma unroll
    for (int nt = 0; nt < 4; ++nt)
#pragma unroll
      for (int r = 0; r < 8; ++r) {
        int row = mB + mt * 16 + r + 8 * hi;
        int col = nB0 + nW + nt * 16 + ln;
        if (F32OUT)
          Cf[(size_t)row * ldc + col] = acc[mt][nt][r];
        else
          Cb[(size_t)row * ldc + col] = f2bf(acc[mt][nt][r]);
      }
}

// ---------------------------------------------------------------------------
// Flash-style causal attention.
// grid = (T/64, B*H); block = 128 (4 waves). Each wave owns a 16-row Q tile.
// qkv: bf16 [B*T, 3072] (q|k|v); vT: bf16 [B,H,64,T]; aout: bf16 [B*T, 1024].
__global__ __launch_bounds__(128) void attn_kernel(
    const unsigned short* __restrict__ qkv,
    const unsigned short* __restrict__ vT,
    unsigned short* __restrict__ aout) {
  __shared__ __align__(16) unsigned short sp[4 * 16 * 40];  // per-wave P staging

  const int lane = threadIdx.x & 31;
  const int w    = threadIdx.x >> 5;
  const int ln   = lane & 15;
  const int hi   = lane >> 4;
  const int bh   = blockIdx.y;
  const int b    = bh >> 4;
  const int h    = bh & 15;
  const int qb   = blockIdx.x * 64 + w * 16;
  const float scale = 0.125f;  // 1/sqrt(64)

  // Q A-fragments for the two K(d)-steps (d 0..31, 32..63)
  const unsigned short* qrow = qkv + (size_t)(b * 2048 + qb + ln) * 3072 + h * 64;
  const int c0 = hi * 16;
  v16bf qa0 = ld_frag(qrow + c0, 8);
  v16bf qa1 = ld_frag(qrow + 32 + c0, 8);

  v8f o[4];
#pragma unroll
  for (int dt = 0; dt < 4; ++dt) o[dt] = (v8f)(0.0f);
  float m[8], l[8];
#pragma unroll
  for (int r = 0; r < 8; ++r) { m[r] = -__builtin_inff(); l[r] = 0.0f; }

  unsigned short* pb = &sp[w * 640];  // 16 rows x 40 shorts (padded stride)

  const int nkb = (qb + 16 + 31) >> 5;  // 32-key blocks covering causal range
  for (int jb = 0; jb < nkb; ++jb) {
    const int kbase = jb * 32;

    // ---- scores S = Q K^T  (two 16-key N-tiles, K-dim = 64 = 2 steps) ----
    v8f s0 = (v8f)(0.0f), s1 = (v8f)(0.0f);
    {
      const unsigned short* kr0 =
          qkv + (size_t)(b * 2048 + kbase + ln) * 3072 + 1024 + h * 64 + hi * 16;
      const unsigned short* kr1 =
          qkv + (size_t)(b * 2048 + kbase + 16 + ln) * 3072 + 1024 + h * 64 + hi * 16;
      s0 = wmma_bf16(qa0, ld_frag(kr0, 8), s0);
      s0 = wmma_bf16(qa1, ld_frag(kr0 + 32, 8), s0);
      s1 = wmma_bf16(qa0, ld_frag(kr1, 8), s1);
      s1 = wmma_bf16(qa1, ld_frag(kr1 + 32, 8), s1);
    }

    // ---- causal mask + online softmax (row = r + 8*hi across 16 lanes) ----
#pragma unroll
    for (int r = 0; r < 8; ++r) {
      const int rowg = qb + r + 8 * hi;
      const int key0 = kbase + ln;
      float a = (key0 <= rowg)      ? s0[r] * scale : -__builtin_inff();
      float c = (key0 + 16 <= rowg) ? s1[r] * scale : -__builtin_inff();
      float mx = fmaxf(a, c);
#pragma unroll
      for (int off = 1; off < 16; off <<= 1)
        mx = fmaxf(mx, __shfl_xor(mx, off, 32));
      const float mn  = fmaxf(m[r], mx);
      const float fac = __expf(m[r] - mn);
      const float p0  = __expf(a - mn);
      const float p1  = __expf(c - mn);
      float sum = p0 + p1;
#pragma unroll
      for (int off = 1; off < 16; off <<= 1) sum += __shfl_xor(sum, off, 32);
      l[r] = l[r] * fac + sum;
      m[r] = mn;
      o[0][r] *= fac; o[1][r] *= fac; o[2][r] *= fac; o[3][r] *= fac;
      const int rl = r + 8 * hi;
      pb[rl * 40 + ln]      = f2bf(p0);
      pb[rl * 40 + ln + 16] = f2bf(p1);
    }
    asm volatile("s_wait_dscnt 0" ::: "memory");

    // ---- O += P V : P(16x32) as A-fragment from LDS; V B-frags from vT ----
    v16bf pa = ld_frag(pb + ln * 40 + hi * 16, 8);
#pragma unroll
    for (int dt = 0; dt < 4; ++dt) {
      const unsigned short* vp =
          vT + (size_t)((b * 16 + h) * 64 + dt * 16 + ln) * 2048 + kbase + hi * 16;
      o[dt] = wmma_bf16(pa, ld_frag(vp, 8), o[dt]);
    }
  }

  // ---- normalize and store [B,T,H*D] ----
#pragma unroll
  for (int dt = 0; dt < 4; ++dt)
#pragma unroll
    for (int r = 0; r < 8; ++r) {
      const int rowg = qb + r + 8 * hi;
      const float val = o[dt][r] / l[r];
      aout[(size_t)(b * 2048 + rowg) * 1024 + h * 64 + dt * 16 + ln] = f2bf(val);
    }
}

// ---------------------------------------------------------------------------
extern "C" void kernel_launch(void* const* d_in, const int* in_sizes, int n_in,
                              void* d_out, int out_size, void* d_ws, size_t ws_size,
                              hipStream_t stream) {
  const float* x      = (const float*)d_in[0];  // [4,2048,1024]
  const float* w_qkv  = (const float*)d_in[1];  // [3072,1024]
  const float* w_proj = (const float*)d_in[2];  // [1024,1024]
  float* outF = (float*)d_out;                  // [4,2048,1024]

  char* ws = (char*)d_ws;  // ~104 MiB used
  unsigned short* xb   = (unsigned short*)(ws);                    // 16 MiB
  unsigned short* wqb  = (unsigned short*)(ws + (16u  << 20));     //  6 MiB
  unsigned short* wpb  = (unsigned short*)(ws + (22u  << 20));     //  2 MiB
  unsigned short* qkvb = (unsigned short*)(ws + (24u  << 20));     // 48 MiB
  unsigned short* vTb  = (unsigned short*)(ws + (72u  << 20));     // 16 MiB
  unsigned short* aob  = (unsigned short*)(ws + (88u  << 20));     // 16 MiB

  const int nx = 4 * 2048 * 1024;      // 8388608
  const int nq = 3 * 1024 * 1024;      // 3145728
  const int np = 1024 * 1024;          // 1048576

  cvt_f32_bf16<<<(nx + 255) / 256, 256, 0, stream>>>(x, xb, nx);
  cvt_f32_bf16<<<(nq + 255) / 256, 256, 0, stream>>>(w_qkv, wqb, nq);
  cvt_f32_bf16<<<(np + 255) / 256, 256, 0, stream>>>(w_proj, wpb, np);

  // qkv = x @ w_qkv^T : M=8192, N=3072, K=1024 -> bf16
  gemm_bf16nt<false><<<dim3(3072 / 128, 8192 / 64), 128, 0, stream>>>(
      xb, wqb, qkvb, nullptr, 1024, 3072, 1024, 1024, 3072);

  vtrans_kernel<<<nx / 256, 256, 0, stream>>>(qkvb, vTb);

  attn_kernel<<<dim3(2048 / 64, 4 * 16), 128, 0, stream>>>(qkvb, vTb, aob);

  // out = attn_out @ w_proj^T : M=8192, N=1024, K=1024 -> f32
  gemm_bf16nt<true><<<dim3(1024 / 128, 8192 / 64), 128, 0, stream>>>(
      aob, wpb, nullptr, outF, 1024, 1024, 1024, 1024, 1024);
}
